// Graphormer_47278999994628
// MI455X (gfx1250) — compile-verified, matrix operand order checked by reading, not run on hardware
//
#include <hip/hip_runtime.h>
#include <hip/hip_bf16.h>

// ---------------------------------------------------------------------------
// Graphormer forward for MI455X (gfx1250, wave32, WMMA + async LDS copies).
// ---------------------------------------------------------------------------

typedef __attribute__((ext_vector_type(16))) _Float16 half16;
typedef __attribute__((ext_vector_type(8)))  _Float16 half8;
typedef __attribute__((ext_vector_type(8)))  float    float8;
typedef __attribute__((ext_vector_type(4)))  int      v4i;

#if defined(__AMDGCN__) && __has_builtin(__builtin_amdgcn_global_load_async_to_lds_b128)
#define USE_ASYNC_LDS 1
typedef __attribute__((address_space(1))) v4i v4i_as1;  // global
typedef __attribute__((address_space(3))) v4i v4i_as3;  // LDS
#else
#define USE_ASYNC_LDS 0
#endif

__device__ __forceinline__ void wait_async_le4() {
#if defined(__AMDGCN__)
#if __has_builtin(__builtin_amdgcn_s_wait_asynccnt)
  __builtin_amdgcn_s_wait_asynccnt(4);
#elif USE_ASYNC_LDS
  asm volatile("s_wait_asynccnt 4" ::: "memory");
#endif
#endif
}
__device__ __forceinline__ void wait_async_0() {
#if defined(__AMDGCN__)
#if __has_builtin(__builtin_amdgcn_s_wait_asynccnt)
  __builtin_amdgcn_s_wait_asynccnt(0);
#elif USE_ASYNC_LDS
  asm volatile("s_wait_asynccnt 0" ::: "memory");
#endif
#endif
}

// 16-byte global -> LDS copy: async on CDNA5 if available, else via VGPRs.
__device__ __forceinline__ void copy16_g2l(const _Float16* src, _Float16* dst) {
#if USE_ASYNC_LDS
  __builtin_amdgcn_global_load_async_to_lds_b128((v4i_as1*)src, (v4i_as3*)dst, 0, 0);
#else
  *(half8*)dst = *(const half8*)src;
#endif
}

__device__ __forceinline__ float8 wmma_f16(half16 a, half16 b, float8 c) {
  return __builtin_amdgcn_wmma_f32_16x16x32_f16(false, a, false, b, (short)0, c, false, false);
}

// Build a 16-half fragment from two contiguous 8-half runs.
// Per CDNA5 ISA: 16-bit A/B fragments hold K runs [8*khalf, 8*khalf+7] and
// [16+8*khalf, 16+8*khalf+7] in elements 0..7 and 8..15 respectively.
__device__ __forceinline__ half16 mk_frag(const _Float16* p0, const _Float16* p1) {
  half8 lo = *(const half8*)p0;
  half8 hi = *(const half8*)p1;
  return __builtin_shufflevector(lo, hi, 0,1,2,3,4,5,6,7,8,9,10,11,12,13,14,15);
}

__device__ __forceinline__ float8 zero8() {
  float8 z;
  #pragma unroll
  for (int i = 0; i < 8; ++i) z[i] = 0.0f;
  return z;
}

// ---------------------------------------------------------------------------
// GEMM: C[M,N] = act(A[M,K] * Wt[N,K]^T + bias)
// A f16 [M,K]; Wt f16 [Npad,K] (pre-transposed, zero-padded rows).
// Block: 256 threads = 8 waves; tile 128x128x32; wave tile 64x32.
// Double-buffered LDS with async global->LDS copies.
// ---------------------------------------------------------------------------
__global__ __launch_bounds__(256) void gemm_kernel(
    const _Float16* __restrict__ A, const _Float16* __restrict__ Wt,
    const float* __restrict__ bias, float* __restrict__ Cf,
    _Float16* __restrict__ Ch, int M, int N, int K, int act)
{
  __shared__ _Float16 LA[2][128][40];  // A tile, row-major [m][k]
  __shared__ _Float16 LW[2][128][40];  // W tile, [n][k]

  const int tid = threadIdx.x;
  const int wv  = tid >> 5;
  const int lane = tid & 31;
  const int lm = lane & 15, kh = lane >> 4;
  const int bm = blockIdx.y * 128, bn = blockIdx.x * 128;
  const int wm = (wv >> 2) * 64, wn = (wv & 3) * 32;

  // per-thread tile-copy coordinates (2 chunks of 8 halfs per array)
  const int m0 = tid >> 2;             // 0..63   (chunk 0 row)
  const int m1 = 64 + (tid >> 2);      // 64..127 (chunk 1 row)
  const int ko0 = (tid & 3) * 8;       // 0,8,16,24

  float8 acc[4][2];
  #pragma unroll
  for (int mi = 0; mi < 4; ++mi)
    #pragma unroll
    for (int ni = 0; ni < 2; ++ni) acc[mi][ni] = zero8();

  const int ntiles = K >> 5;

  auto issue = [&](int t, int buf) {
    const int kt = t << 5;
    copy16_g2l(A  + (size_t)(bm + m0) * K + kt + ko0, &LA[buf][m0][ko0]);
    copy16_g2l(A  + (size_t)(bm + m1) * K + kt + ko0, &LA[buf][m1][ko0]);
    copy16_g2l(Wt + (size_t)(bn + m0) * K + kt + ko0, &LW[buf][m0][ko0]);
    copy16_g2l(Wt + (size_t)(bn + m1) * K + kt + ko0, &LW[buf][m1][ko0]);
  };

  issue(0, 0);
  for (int t = 0; t < ntiles; ++t) {
    const int cur = t & 1;
    __syncthreads();                       // prev compute done -> safe to fill cur^1
    if (t + 1 < ntiles) { issue(t + 1, cur ^ 1); wait_async_le4(); }
    else                { wait_async_0(); }
    __syncthreads();                       // tile `cur` visible to all waves

    half16 af[4], bf[2];
    #pragma unroll
    for (int mi = 0; mi < 4; ++mi) {
      const _Float16* r = &LA[cur][wm + mi * 16 + lm][0];
      af[mi] = mk_frag(r + 8 * kh, r + 16 + 8 * kh);
    }
    #pragma unroll
    for (int ni = 0; ni < 2; ++ni) {
      const _Float16* r = &LW[cur][wn + ni * 16 + lm][0];
      bf[ni] = mk_frag(r + 8 * kh, r + 16 + 8 * kh);
    }
    #pragma unroll
    for (int mi = 0; mi < 4; ++mi)
      #pragma unroll
      for (int ni = 0; ni < 2; ++ni)
        acc[mi][ni] = wmma_f16(af[mi], bf[ni], acc[mi][ni]);
  }

  // Epilogue: C/D fragment layout: element v -> row v + 8*kh, col = lane&15
  #pragma unroll
  for (int mi = 0; mi < 4; ++mi)
    #pragma unroll
    for (int ni = 0; ni < 2; ++ni)
      #pragma unroll
      for (int v = 0; v < 8; ++v) {
        int row = bm + wm + mi * 16 + v + 8 * kh;
        int col = bn + wn + ni * 16 + lm;
        if (col < N) {
          float cv = acc[mi][ni][v] + (bias ? bias[col] : 0.0f);
          if (act == 1) cv = fmaxf(cv, 0.0f);                    // ReLU
          else if (act == 2) cv = (cv >= 0.0f) ? cv : 0.1f * cv; // LeakyReLU(0.1)
          size_t o = (size_t)row * N + col;
          if (Cf) Cf[o] = cv;
          if (Ch) Ch[o] = (_Float16)cv;
        }
      }
}

// ---------------------------------------------------------------------------
// Flash attention: one wave per 16-row q tile, online softmax over 32-key tiles.
// Q,K f16 [B,N,H,dh]; Vt f16 [B,H,dh,N]; BM f16 [B,H,N,N] masked bias.
// Out O f16 [B,N,H,dh].
// ---------------------------------------------------------------------------
#define AN 1024
#define AH 4
#define ADH 128

__global__ __launch_bounds__(256) void attn_kernel(
    const _Float16* __restrict__ Q, const _Float16* __restrict__ Kk,
    const _Float16* __restrict__ Vt, const _Float16* __restrict__ BM,
    _Float16* __restrict__ O)
{
  __shared__ _Float16 Pl[8][16][40];  // per-wave P staging (C-layout -> A-layout)

  const int tid = threadIdx.x;
  const int wv = tid >> 5;
  const int lane = tid & 31;
  const int lm = lane & 15, kh = lane >> 4;
  const int b = blockIdx.z, h = blockIdx.y;
  const int q0 = (blockIdx.x * 8 + wv) * 16;
  const size_t bh = (size_t)b * AH + h;
  const float scale = 0.08838834764831845f;  // 1/sqrt(128)

  half16 qf[4];
  {
    const _Float16* qrow = Q + (((size_t)b * AN + (q0 + lm)) * AH + h) * ADH;
    #pragma unroll
    for (int kk = 0; kk < 4; ++kk)
      qf[kk] = mk_frag(qrow + 32 * kk + 8 * kh, qrow + 32 * kk + 16 + 8 * kh);
  }

  float8 Of[8];
  #pragma unroll
  for (int nt = 0; nt < 8; ++nt) Of[nt] = zero8();
  float rmax[8], rsum[8];
  #pragma unroll
  for (int v = 0; v < 8; ++v) { rmax[v] = -3.0e38f; rsum[v] = 0.0f; }

  const _Float16* bmrow = BM + (bh * AN + (q0 + lm)) * AN;

  for (int kt = 0; kt < AN; kt += 32) {
    if (kt + 32 < AN) __builtin_prefetch(bmrow + kt + 32, 0, 1);

    // --- S = scale*(Q K^T) + bias over 32 keys (two 16-col C fragments) ---
    float8 Sc[2];
    #pragma unroll
    for (int c = 0; c < 2; ++c) {
      Sc[c] = zero8();
      int key = kt + c * 16 + lm;
      const _Float16* krow = Kk + (((size_t)b * AN + key) * AH + h) * ADH;
      #pragma unroll
      for (int kk = 0; kk < 4; ++kk) {
        half16 bf = mk_frag(krow + 32 * kk + 8 * kh, krow + 32 * kk + 16 + 8 * kh);
        Sc[c] = wmma_f16(qf[kk], bf, Sc[c]);
      }
    }
    #pragma unroll
    for (int c = 0; c < 2; ++c)
      #pragma unroll
      for (int v = 0; v < 8; ++v) {
        int r = v + 8 * kh;
        int key = kt + c * 16 + lm;
        float bmv = (float)BM[(bh * AN + (q0 + r)) * AN + key];
        Sc[c][v] = Sc[c][v] * scale + bmv;
      }

    // --- online softmax: row-wise max/sum via 16-lane shuffles ---
    float mnew[8], alpha[8];
    #pragma unroll
    for (int v = 0; v < 8; ++v) {
      float cnd = fmaxf(Sc[0][v], Sc[1][v]);
      #pragma unroll
      for (int m = 1; m < 16; m <<= 1) cnd = fmaxf(cnd, __shfl_xor(cnd, m, 32));
      mnew[v] = fmaxf(rmax[v], cnd);
      alpha[v] = __expf(rmax[v] - mnew[v]);
      rmax[v] = mnew[v];
    }
    #pragma unroll
    for (int v = 0; v < 8; ++v) {
      float p0 = __expf(Sc[0][v] - mnew[v]);
      float p1 = __expf(Sc[1][v] - mnew[v]);
      Pl[wv][v + 8 * kh][lm]      = (_Float16)p0;
      Pl[wv][v + 8 * kh][16 + lm] = (_Float16)p1;
      float ps = p0 + p1;
      #pragma unroll
      for (int m = 1; m < 16; m <<= 1) ps += __shfl_xor(ps, m, 32);
      rsum[v] = rsum[v] * alpha[v] + ps;
    }
    #pragma unroll
    for (int nt = 0; nt < 8; ++nt)
      #pragma unroll
      for (int v = 0; v < 8; ++v) Of[nt][v] *= alpha[v];

    asm volatile("s_wait_dscnt 0" ::: "memory");  // P writes -> cross-lane reads

    // --- O += P @ V: A-frag from LDS P, B-frags contiguous from Vt ---
    half16 pf = mk_frag(&Pl[wv][lm][8 * kh], &Pl[wv][lm][16 + 8 * kh]);
    const _Float16* vbase = Vt + bh * ADH * (size_t)AN;
    #pragma unroll
    for (int nt = 0; nt < 8; ++nt) {
      const _Float16* vrow = vbase + (size_t)(nt * 16 + lm) * AN + kt;
      half16 vf = mk_frag(vrow + 8 * kh, vrow + 16 + 8 * kh);
      Of[nt] = wmma_f16(pf, vf, Of[nt]);
    }
  }

  #pragma unroll
  for (int nt = 0; nt < 8; ++nt)
    #pragma unroll
    for (int v = 0; v < 8; ++v) {
      int r = v + 8 * kh;
      float o = Of[nt][v] / rsum[v];
      O[(((size_t)b * AN + (q0 + r)) * AH + h) * ADH + nt * 16 + lm] = (_Float16)o;
    }
}

// ---------------------------------------------------------------------------
// LayerNorm(res + add) * g + b  ->  xo (f32) and xh (f16). One wave per row.
// ---------------------------------------------------------------------------
__global__ __launch_bounds__(256) void ln_kernel(
    const float* __restrict__ res, const float* __restrict__ add,
    const float* __restrict__ g, const float* __restrict__ bb,
    float* __restrict__ xo, _Float16* __restrict__ xh)
{
  const int row = blockIdx.x * 8 + (threadIdx.x >> 5);
  const int lane = threadIdx.x & 31;
  const size_t base = (size_t)row * 512;
  float vbuf[16];
  float s = 0.0f, ss = 0.0f;
  #pragma unroll
  for (int i = 0; i < 16; ++i) {
    int c = lane + i * 32;
    float v = res[base + c] + add[base + c];
    vbuf[i] = v; s += v; ss += v * v;
  }
  #pragma unroll
  for (int m = 1; m < 32; m <<= 1) { s += __shfl_xor(s, m, 32); ss += __shfl_xor(ss, m, 32); }
  float mean = s * (1.0f / 512.0f);
  float var = ss * (1.0f / 512.0f) - mean * mean;
  float rstd = rsqrtf(var + 1e-5f);
  #pragma unroll
  for (int i = 0; i < 16; ++i) {
    int c = lane + i * 32;
    float o = (vbuf[i] - mean) * rstd * g[c] + bb[c];
    xo[base + c] = o;
    xh[base + c] = (_Float16)o;
  }
}

// ---------------------------------------------------------------------------
// Elementwise helpers
// ---------------------------------------------------------------------------
__global__ void cvt_f32_f16_kernel(const float* __restrict__ in, _Float16* __restrict__ out, long n) {
  long i = (long)blockIdx.x * blockDim.x + threadIdx.x;
  if (i < n) out[i] = (_Float16)in[i];
}

// Wt[n,k] = (n < N) ? (f16)W[k,n] : 0   for n in [0, Npad)
__global__ void wprep_kernel(const float* __restrict__ W, _Float16* __restrict__ Wt,
                             int K, int N, long total) {
  long idx = (long)blockIdx.x * blockDim.x + threadIdx.x;
  if (idx >= total) return;
  int k = (int)(idx % K);
  int n = (int)(idx / K);
  Wt[idx] = (n < N) ? (_Float16)W[(size_t)k * N + n] : (_Float16)0.0f;
}

// bm[b,h,q,k] = mask[b,q,k] ? -30000 : bias[b,q,k,h]
__global__ void bm16_kernel(const float* __restrict__ bias, const int* __restrict__ mask,
                            _Float16* __restrict__ bm) {
  long idx = (long)blockIdx.x * blockDim.x + threadIdx.x;  // 8*4*1024*1024
  int k = idx & 1023;
  int q = (idx >> 10) & 1023;
  int h = (idx >> 20) & 3;
  int b = (int)(idx >> 22);
  int mk = mask[((size_t)b << 20) | ((size_t)q << 10) | k];
  float bv = bias[(((size_t)b * 1024 + q) * 1024 + k) * 4 + h];
  bm[idx] = (_Float16)(mk ? -30000.0f : bv);
}

// x = h + deg_in_emb[clip(in)] + deg_out_emb[clip(out)];  writes f32 + f16
__global__ void degadd_kernel(const float* __restrict__ hb, const int* __restrict__ ind,
                              const int* __restrict__ outd, const float* __restrict__ ein,
                              const float* __restrict__ eout, float* __restrict__ x,
                              _Float16* __restrict__ xh) {
  long idx = (long)blockIdx.x * blockDim.x + threadIdx.x;  // 8192*512
  int d = idx & 511;
  int row = (int)(idx >> 9);
  int di = ind[row];  di = di < 0 ? 0 : (di > 8 ? 8 : di);
  int dq = outd[row]; dq = dq < 0 ? 0 : (dq > 8 ? 8 : dq);
  float v = hb[idx] + ein[(size_t)di * 512 + d] + eout[(size_t)dq * 512 + d];
  x[idx] = v;
  xh[idx] = (_Float16)v;
}

// Vt[b,h,d,n] = Vh[b,n,h,d]
__global__ void vt_kernel(const _Float16* __restrict__ Vh, _Float16* __restrict__ Vt) {
  long idx = (long)blockIdx.x * blockDim.x + threadIdx.x;  // 8*4*128*1024
  int n = idx & 1023;
  int d = (idx >> 10) & 127;
  int h = (idx >> 17) & 3;
  int b = (int)(idx >> 19);
  Vt[idx] = Vh[(((size_t)b * 1024 + n) * 4 + h) * 128 + d];
}

// sel[b*128+p, d] = xh[b, po[b,p], d]
__global__ void gather_kernel(const _Float16* __restrict__ xh, const int* __restrict__ po,
                              _Float16* __restrict__ sel) {
  long idx = (long)blockIdx.x * blockDim.x + threadIdx.x;  // 1024*512
  int d = idx & 511;
  int i = (int)(idx >> 9);
  int b = i >> 7;
  int p = i & 127;
  int node = po[b * 128 + p];
  sel[idx] = xh[(((size_t)b * 1024 + node) * 512) + d];
}

// ---------------------------------------------------------------------------
// Launcher
// ---------------------------------------------------------------------------
extern "C" void kernel_launch(void* const* d_in, const int* in_sizes, int n_in,
                              void* d_out, int out_size, void* d_ws, size_t ws_size,
                              hipStream_t stream) {
  (void)in_sizes; (void)n_in; (void)out_size; (void)ws_size;

  const float* node_feat = (const float*)d_in[0];
  const int*   in_deg    = (const int*)d_in[1];
  const int*   out_deg   = (const int*)d_in[2];
  const int*   mask      = (const int*)d_in[4];
  const int*   po_idx    = (const int*)d_in[5];
  const float* bias      = (const float*)d_in[6];
  const float* ein       = (const float*)d_in[7];
  const float* eout      = (const float*)d_in[8];
  const float* mnW1 = (const float*)d_in[9];  const float* mnb1 = (const float*)d_in[10];
  const float* mnW2 = (const float*)d_in[11]; const float* mnb2 = (const float*)d_in[12];
  const float* Wq = (const float*)d_in[13]; const float* bq = (const float*)d_in[14];
  const float* Wk = (const float*)d_in[15]; const float* bk = (const float*)d_in[16];
  const float* Wv = (const float*)d_in[17]; const float* bv = (const float*)d_in[18];
  const float* Wo = (const float*)d_in[19]; const float* bo = (const float*)d_in[20];
  const float* ln1g = (const float*)d_in[21]; const float* ln1b = (const float*)d_in[22];
  const float* W1 = (const float*)d_in[23]; const float* b1 = (const float*)d_in[24];
  const float* W2 = (const float*)d_in[25]; const float* b2 = (const float*)d_in[26];
  const float* ln2g = (const float*)d_in[27]; const float* ln2b = (const float*)d_in[28];
  const float* moW1 = (const float*)d_in[29]; const float* mob1 = (const float*)d_in[30];
  const float* moW2 = (const float*)d_in[31]; const float* mob2 = (const float*)d_in[32];

  const int M = 8192, D = 512;

  // workspace bump allocator (256B aligned)
  char* wsp = (char*)d_ws;
  auto alloc = [&](size_t bytes) -> void* {
    void* p = (void*)wsp;
    wsp += (bytes + 255) & ~(size_t)255;
    return p;
  };
  float*     XF32 = (float*)alloc((size_t)M * D * 4);
  _Float16*  XH   = (_Float16*)alloc((size_t)M * D * 2);
  _Float16*  BM16 = (_Float16*)alloc((size_t)8 * 4 * 1024 * 1024 * 2);
  _Float16*  QH   = (_Float16*)alloc((size_t)M * D * 2);
  _Float16*  KH   = (_Float16*)alloc((size_t)M * D * 2);
  _Float16*  VH   = (_Float16*)alloc((size_t)M * D * 2);
  _Float16*  VT   = (_Float16*)alloc((size_t)M * D * 2);
  _Float16*  OH   = (_Float16*)alloc((size_t)M * D * 2);
  float*     PROJ = (float*)alloc((size_t)M * D * 4);
  _Float16*  FFNH = (_Float16*)alloc((size_t)M * 1024 * 2);
  _Float16*  T1H  = (_Float16*)alloc((size_t)M * 256 * 2);
  _Float16*  NODEH= (_Float16*)alloc((size_t)M * 64 * 2);
  _Float16*  SELH = (_Float16*)alloc((size_t)1024 * 512 * 2);
  _Float16*  MO1H = (_Float16*)alloc((size_t)1024 * 256 * 2);

  // weight prep: f32 [K,N] -> f16 [Npad,K] transposed, zero-padded
  auto prep = [&](const float* W, int K_, int N_) -> _Float16* {
    int Npad = (N_ + 127) & ~127;
    long total = (long)Npad * K_;
    _Float16* Wt = (_Float16*)alloc((size_t)total * 2);
    wprep_kernel<<<(total + 255) / 256, 256, 0, stream>>>(W, Wt, K_, N_, total);
    return Wt;
  };

  _Float16* mnW1t = prep(mnW1, 64, 256);
  _Float16* mnW2t = prep(mnW2, 256, 512);
  _Float16 *WqT[6], *WkT[6], *WvT[6], *WoT[6], *W1T[6], *W2T[6];
  for (int i = 0; i < 6; ++i) {
    WqT[i] = prep(Wq + (size_t)i * 512 * 512, 512, 512);
    WkT[i] = prep(Wk + (size_t)i * 512 * 512, 512, 512);
    WvT[i] = prep(Wv + (size_t)i * 512 * 512, 512, 512);
    WoT[i] = prep(Wo + (size_t)i * 512 * 512, 512, 512);
    W1T[i] = prep(W1 + (size_t)i * 512 * 1024, 512, 1024);
    W2T[i] = prep(W2 + (size_t)i * 1024 * 512, 1024, 512);
  }
  _Float16* moW1t = prep(moW1, 512, 256);
  _Float16* moW2t = prep(moW2, 256, 1);

  auto gemm = [&](const _Float16* A, const _Float16* Wt, const float* bia,
                  float* Cf, _Float16* Ch, int M_, int N_, int K_, int act) {
    dim3 g((N_ + 127) / 128, M_ / 128);
    gemm_kernel<<<g, 256, 0, stream>>>(A, Wt, bia, Cf, Ch, M_, N_, K_, act);
  };

  // masked bias precompute (dominant HBM traffic: f32+int -> f16 once)
  bm16_kernel<<<(8l * 4 * 1024 * 1024) / 256, 256, 0, stream>>>(bias, mask, BM16);

  // node MLP: Linear(64->256) + LeakyReLU(0.1) + Linear(256->512)
  cvt_f32_f16_kernel<<<((long)M * 64 + 255) / 256, 256, 0, stream>>>(node_feat, NODEH, (long)M * 64);
  gemm(NODEH, mnW1t, mnb1, nullptr, T1H, M, 256, 64, 2);
  gemm(T1H, mnW2t, mnb2, PROJ, nullptr, M, 512, 256, 0);
  degadd_kernel<<<((long)M * 512) / 256, 256, 0, stream>>>(PROJ, in_deg, out_deg, ein, eout, XF32, XH);

  for (int i = 0; i < 6; ++i) {
    gemm(XH, WqT[i], bq + (size_t)i * 512, nullptr, QH, M, 512, 512, 0);
    gemm(XH, WkT[i], bk + (size_t)i * 512, nullptr, KH, M, 512, 512, 0);
    gemm(XH, WvT[i], bv + (size_t)i * 512, nullptr, VH, M, 512, 512, 0);
    vt_kernel<<<((long)8 * 4 * 128 * 1024) / 256, 256, 0, stream>>>(VH, VT);

    attn_kernel<<<dim3(8, 4, 8), 256, 0, stream>>>(QH, KH, VT, BM16, OH);

    gemm(OH, WoT[i], bo + (size_t)i * 512, PROJ, nullptr, M, 512, 512, 0);
    ln_kernel<<<M / 8, 256, 0, stream>>>(XF32, PROJ, ln1g + (size_t)i * 512, ln1b + (size_t)i * 512, XF32, XH);

    gemm(XH, W1T[i], b1 + (size_t)i * 1024, nullptr, FFNH, M, 1024, 512, 1);   // ReLU
    gemm(FFNH, W2T[i], b2 + (size_t)i * 512, PROJ, nullptr, M, 512, 1024, 0);
    ln_kernel<<<M / 8, 256, 0, stream>>>(XF32, PROJ, ln2g + (size_t)i * 512, ln2b + (size_t)i * 512, XF32, XH);
  }

  // gather PO nodes, output MLP: Linear(512->256)+LeakyReLU, Linear(256->1)
  gather_kernel<<<((long)1024 * 512) / 256, 256, 0, stream>>>(XH, po_idx, SELH);
  gemm(SELH, moW1t, mob1, nullptr, MO1H, 1024, 256, 512, 2);
  gemm(MO1H, moW2t, mob2, (float*)d_out, nullptr, 1024, 1, 256, 0);
}
